// GroupQueryAtention_7627861918062
// MI455X (gfx1250) — compile-verified, matrix operand order checked by reading, not run on hardware
//
#include <hip/hip_runtime.h>

typedef __attribute__((ext_vector_type(16))) _Float16 v16h;
typedef __attribute__((ext_vector_type(8)))  _Float16 h8;
typedef __attribute__((ext_vector_type(8)))  float    v8f;
typedef __attribute__((ext_vector_type(4)))  float    f4;
typedef __attribute__((ext_vector_type(4)))  unsigned int u32x4;
typedef __attribute__((ext_vector_type(8)))  int      i32x8;
typedef __attribute__((ext_vector_type(4)))  int      i32x4;
typedef int v4i __attribute__((vector_size(16)));   // exact type the async builtin wants

typedef __attribute__((address_space(3))) _Float16 lds_f16;

#if __has_builtin(__builtin_amdgcn_global_load_async_to_lds_b128)
#define HAVE_ASYNC_LDS 1
#else
#define HAVE_ASYNC_LDS 0
#endif

#if __has_builtin(__builtin_amdgcn_tensor_load_to_lds) && \
    __has_builtin(__builtin_amdgcn_s_wait_tensorcnt)
#define HAVE_TDM 1
#else
#define HAVE_TDM 0
#endif

// 16-byte global -> LDS copy: async (ASYNCcnt-tracked) when available.
__device__ __forceinline__ void copy_b128(const _Float16* gsrc, _Float16* ldst) {
#if HAVE_ASYNC_LDS
  __builtin_amdgcn_global_load_async_to_lds_b128((v4i*)gsrc, (v4i*)ldst, 0, 0);
#else
  *(h8*)ldst = *(const h8*)gsrc;
#endif
}

__device__ __forceinline__ void wait_async_lds() {
#if HAVE_ASYNC_LDS
#if __has_builtin(__builtin_amdgcn_s_wait_asynccnt)
  __builtin_amdgcn_s_wait_asynccnt(0);
#else
  asm volatile("s_wait_asynccnt 0x0" ::: "memory");
#endif
#endif
}

#if HAVE_TDM
// TDM: move one [32 rows x 128 halves] tile (row stride 512 halves) into LDS,
// padding 4 DWORDs after every 64 DWORDs so rows land at stride 136 halves.
__device__ __forceinline__ void tdm_load_k_tile(const _Float16* gsrc, unsigned lds_byte_off) {
  const size_t ga = (size_t)gsrc;
  u32x4 g0;
  g0[0] = 1u;                                                   // count=1, user mode
  g0[1] = lds_byte_off;                                         // lds_addr
  g0[2] = (unsigned)(ga & 0xFFFFFFFFull);                       // global_addr[31:0]
  g0[3] = (unsigned)((ga >> 32) & 0x01FFFFFFull) | (2u << 30);  // addr[56:32] | type=2
  i32x8 g1;
  // mask=0 | data_size=1(2B) | pad_enable | pad_interval=5(64dw) | pad_amount=3(4dw)
  g1[0] = (1 << 16) | (1 << 20) | (5 << 22) | (3 << 25);
  g1[1] = (int)(512u << 16);   // tensor_dim0 = 512 (bits 79:48, low half)
  g1[2] = (int)(32u << 16);    // tensor_dim0 hi=0 | tensor_dim1 = 32 (bits 111:80)
  g1[3] = (int)(128u << 16);   // tensor_dim1 hi=0 | tile_dim0 = 128 (bits 127:112)
  g1[4] = 32;                  // tile_dim1 = 32, tile_dim2 = 0 (2D tile)
  g1[5] = 512;                 // tensor_dim0_stride = 512 (bits 207:160, low word)
  g1[6] = 0;                   // stride0 hi | tensor_dim1_stride lo (unused)
  g1[7] = 0;
  i32x4 z4 = {};
#if __clang_major__ >= 23
  i32x8 z8 = {};
  __builtin_amdgcn_tensor_load_to_lds(g0, g1, z4, z4, z8, 0);
#else
  __builtin_amdgcn_tensor_load_to_lds(g0, g1, z4, z4, 0);
#endif
}
#endif

// Load one 16x32 f16 WMMA operand fragment (A- or B-matrix) from an LDS tile
// stored [row][K] with leading dimension `ld` (halves). Matches the CDNA5
// 16-bit operand layout: lanes 0-15 -> K {0..7,16..23}+koff, lanes 16-31 ->
// K {8..15,24..31}+koff, row = row0 + (lane&15).
__device__ __forceinline__ v16h load_frag16(const _Float16* tile, int row0, int ld,
                                            int koff, int lane) {
  const _Float16* p = tile + (size_t)(row0 + (lane & 15)) * ld + koff + ((lane >> 4) << 3);
  h8 lo = *(const h8*)p;
  h8 hi = *(const h8*)(p + 16);
  v16h r;
#pragma unroll
  for (int i = 0; i < 8; ++i) { r[i] = lo[i]; r[8 + i] = hi[i]; }
  return r;
}

__device__ __forceinline__ v8f wmma_f16(v16h a, v16h b, v8f c) {
  return __builtin_amdgcn_wmma_f32_16x16x32_f16(false, a, false, b, (short)0, c,
                                                false, false);
}

// Y[M,N] = A[M,K] @ W[K,N] + bias.  A is f32 or f16; Y is f16 (workspace) or f32.
// 128x128 tile per 256-thread block (8 waves, 32x64 per wave => 8 WMMAs/k-step).
template <bool A_HALF, bool OUT_HALF>
__global__ __launch_bounds__(256) void gemm_bias_kernel(
    const void* __restrict__ Ap, const float* __restrict__ W,
    const float* __restrict__ bias, void* __restrict__ Yp, int M, int N, int K) {
  __shared__ _Float16 sA[128][40];  // [m][k], padded
  __shared__ _Float16 sB[128][40];  // [n][k] (transposed on load), padded

  const int t    = threadIdx.x;
  const int lane = t & 31;
  const int w    = t >> 5;
  const int m0   = blockIdx.y * 128;
  const int n0   = blockIdx.x * 128;
  const int wr   = (w >> 1) * 32;
  const int wc   = (w & 1) * 64;

  v8f acc[2][4];
#pragma unroll
  for (int i = 0; i < 2; ++i)
#pragma unroll
    for (int j = 0; j < 4; ++j) acc[i][j] = {};

  for (int k0 = 0; k0 < K; k0 += 32) {
    // ---- stage A tile: 128 rows x 32 K (2 threads per row) ----
    {
      const int m  = t >> 1;
      const int kh = (t & 1) * 16;
      _Float16* dst = &sA[m][kh];
      if constexpr (A_HALF) {
        const _Float16* src = (const _Float16*)Ap + (size_t)(m0 + m) * K + k0 + kh;
        copy_b128(src, dst);          // async global->LDS when available
        copy_b128(src + 8, dst + 8);
      } else {
        const float* src = (const float*)Ap + (size_t)(m0 + m) * K + k0 + kh;
        f4 x0 = ((const f4*)src)[0];
        f4 x1 = ((const f4*)src)[1];
        f4 x2 = ((const f4*)src)[2];
        f4 x3 = ((const f4*)src)[3];
        h8 h0, h1;
#pragma unroll
        for (int i = 0; i < 4; ++i) {
          h0[i] = (_Float16)x0[i]; h0[4 + i] = (_Float16)x1[i];
          h1[i] = (_Float16)x2[i]; h1[4 + i] = (_Float16)x3[i];
        }
        *(h8*)dst       = h0;
        *(h8*)(dst + 8) = h1;
      }
    }
    // ---- stage W tile transposed: sB[n][k] (coalesced global reads) ----
    {
      const int n = t & 127;
#pragma unroll
      for (int it = 0; it < 16; ++it) {
        const int k = it * 2 + (t >> 7);
        sB[n][k] = (_Float16)W[(size_t)(k0 + k) * N + n0 + n];
      }
      if (k0 + 32 < K)  // prefetch next-k W tile (global_prefetch_b8)
        __builtin_prefetch(&W[(size_t)(k0 + 32 + (t >> 7)) * N + n0 + n], 0, 1);
    }
    if constexpr (A_HALF) wait_async_lds();
    __syncthreads();

    v16h af[2], bf[4];
#pragma unroll
    for (int mi = 0; mi < 2; ++mi) af[mi] = load_frag16(&sA[0][0], wr + mi * 16, 40, 0, lane);
#pragma unroll
    for (int ni = 0; ni < 4; ++ni) bf[ni] = load_frag16(&sB[0][0], wc + ni * 16, 40, 0, lane);
#pragma unroll
    for (int mi = 0; mi < 2; ++mi)
#pragma unroll
      for (int ni = 0; ni < 4; ++ni) acc[mi][ni] = wmma_f16(af[mi], bf[ni], acc[mi][ni]);
    __syncthreads();
  }

  // ---- epilogue: bias + store (C layout: row = (lane>>4)*8 + r, col = lane&15) ----
#pragma unroll
  for (int ni = 0; ni < 4; ++ni) {
    const int col = n0 + wc + ni * 16 + (lane & 15);
    const float bvl = bias[col];
#pragma unroll
    for (int mi = 0; mi < 2; ++mi) {
#pragma unroll
      for (int r = 0; r < 8; ++r) {
        const int row = m0 + wr + mi * 16 + ((lane >> 4) << 3) + r;
        const float v = acc[mi][ni][r] + bvl;
        if constexpr (OUT_HALF)
          ((_Float16*)Yp)[(size_t)row * N + col] = (_Float16)v;
        else
          ((float*)Yp)[(size_t)row * N + col] = v;
      }
    }
  }
}

// Flash-style GQA attention, NO 1/sqrt(D) scaling (faithful to reference).
// grid.x = S/128 query blocks, grid.y = B*Hq heads. 8 waves, 16 q-rows each.
__global__ __launch_bounds__(256) void gqa_attn_kernel(
    const _Float16* __restrict__ Qh, const _Float16* __restrict__ Kh,
    const _Float16* __restrict__ Vh, _Float16* __restrict__ Ctx, int S) {
  constexpr int D  = 128;
  constexpr int BC = 32;  // keys per block
  __shared__ _Float16 sK[BC][136];    // [key][d]   (B operand for QK^T)
  __shared__ _Float16 sVt[D][40];     // [d][key]   (B operand for P*V)
  __shared__ _Float16 sP[8][16][40];  // per-wave P staging (C->A layout fix)

  const int t    = threadIdx.x;
  const int lane = t & 31;
  const int w    = t >> 5;
  const int b    = blockIdx.y >> 4;
  const int h    = blockIdx.y & 15;
  const int kvh  = h >> 2;
  const int qr0  = blockIdx.x * 128 + w * 16;

  const size_t qStride = 2048;  // Hq*D
  const size_t kStride = 512;   // Hkv*D
  const _Float16* Qb = Qh + (size_t)b * S * qStride + (size_t)h * D;
  const _Float16* Kb = Kh + (size_t)b * S * kStride + (size_t)kvh * D;
  const _Float16* Vb = Vh + (size_t)b * S * kStride + (size_t)kvh * D;

#if HAVE_TDM
  const unsigned ldsK_off = (unsigned)(size_t)(lds_f16*)&sK[0][0];
#endif

  // Q fragments live in registers for the whole pass (4 d-steps of 32).
  v16h qf[4];
#pragma unroll
  for (int ds = 0; ds < 4; ++ds) {
    const _Float16* p =
        Qb + (size_t)(qr0 + (lane & 15)) * qStride + ds * 32 + ((lane >> 4) << 3);
    h8 lo = *(const h8*)p;
    h8 hi = *(const h8*)(p + 16);
#pragma unroll
    for (int i = 0; i < 8; ++i) { qf[ds][i] = lo[i]; qf[ds][8 + i] = hi[i]; }
  }

  v8f o[8];
#pragma unroll
  for (int i = 0; i < 8; ++i) o[i] = {};
  float mrow[8], lrow[8];
#pragma unroll
  for (int r = 0; r < 8; ++r) { mrow[r] = -1e30f; lrow[r] = 0.f; }

  for (int kb = 0; kb < S; kb += BC) {
    __syncthreads();  // previous iteration's LDS consumers done
    const int key = t >> 3;
    const int d0  = (t & 7) * 16;
#if HAVE_TDM
    if (w == 0) {  // one TDM descriptor moves the whole padded K tile
      tdm_load_k_tile(Kb + (size_t)kb * kStride, ldsK_off);
      __builtin_amdgcn_s_wait_tensorcnt(0);
    }
#else
    {
      const _Float16* ks = Kb + (size_t)(kb + key) * kStride + d0;
      copy_b128(ks, &sK[key][d0]);
      copy_b128(ks + 8, &sK[key][d0 + 8]);
    }
#endif
    {  // V tile transposed into LDS: [d][key]
      const _Float16* vs = Vb + (size_t)(kb + key) * kStride + d0;
      h8 v0 = *(const h8*)vs;
      h8 v1 = *(const h8*)(vs + 8);
#pragma unroll
      for (int i = 0; i < 8; ++i) {
        sVt[d0 + i][key]     = v0[i];
        sVt[d0 + 8 + i][key] = v1[i];
      }
    }
#if !HAVE_TDM
    wait_async_lds();
#endif
    __syncthreads();

    // ---- scores S[16][32] = Q @ K^T (8 WMMAs) ----
    v8f sc[2];
#pragma unroll
    for (int nb = 0; nb < 2; ++nb) {
      sc[nb] = {};
#pragma unroll
      for (int ds = 0; ds < 4; ++ds) {
        v16h kf = load_frag16(&sK[0][0], nb * 16, 136, ds * 32, lane);
        sc[nb] = wmma_f16(qf[ds], kf, sc[nb]);
      }
    }

    // ---- online softmax (row reductions across the 16-lane halves) ----
    float bm[8];
#pragma unroll
    for (int r = 0; r < 8; ++r) bm[r] = fmaxf(sc[0][r], sc[1][r]);
#pragma unroll
    for (int off = 1; off < 16; off <<= 1)
#pragma unroll
      for (int r = 0; r < 8; ++r) bm[r] = fmaxf(bm[r], __shfl_xor(bm[r], off, 32));

    float alpha[8], ps[8];
#pragma unroll
    for (int r = 0; r < 8; ++r) {
      const float mn = fmaxf(mrow[r], bm[r]);
      alpha[r] = __expf(mrow[r] - mn);
      mrow[r]  = mn;
      ps[r]    = 0.f;
    }
#pragma unroll
    for (int nb = 0; nb < 2; ++nb)
#pragma unroll
      for (int r = 0; r < 8; ++r) {
        const float p = __expf(sc[nb][r] - mrow[r]);
        ps[r] += p;
        sP[w][((lane >> 4) << 3) + r][nb * 16 + (lane & 15)] = (_Float16)p;
      }
#pragma unroll
    for (int off = 1; off < 16; off <<= 1)
#pragma unroll
      for (int r = 0; r < 8; ++r) ps[r] += __shfl_xor(ps[r], off, 32);
#pragma unroll
    for (int r = 0; r < 8; ++r) lrow[r] = alpha[r] * lrow[r] + ps[r];

    // ---- rescale O, then O += P @ V (8 WMMAs) ----
#pragma unroll
    for (int ob = 0; ob < 8; ++ob)
#pragma unroll
      for (int r = 0; r < 8; ++r) o[ob][r] *= alpha[r];

    v16h pf = load_frag16(&sP[w][0][0], 0, 40, 0, lane);
#pragma unroll
    for (int ob = 0; ob < 8; ++ob) {
      v16h vf = load_frag16(&sVt[0][0], ob * 16, 40, 0, lane);
      o[ob] = wmma_f16(pf, vf, o[ob]);
    }
  }

  // ---- normalize and emit context in [B,S,H] head-sliced layout ----
  _Float16* out = Ctx + (size_t)b * S * qStride + (size_t)h * D;
#pragma unroll
  for (int r = 0; r < 8; ++r) {
    const float inv = 1.f / lrow[r];
    const int row   = qr0 + ((lane >> 4) << 3) + r;
#pragma unroll
    for (int ob = 0; ob < 8; ++ob) {
      const int col = ob * 16 + (lane & 15);
      out[(size_t)row * qStride + col] = (_Float16)(o[ob][r] * inv);
    }
  }
}

extern "C" void kernel_launch(void* const* d_in, const int* in_sizes, int n_in,
                              void* d_out, int out_size, void* d_ws, size_t ws_size,
                              hipStream_t stream) {
  const float* query = (const float*)d_in[0];
  const float* key_  = (const float*)d_in[1];
  const float* value = (const float*)d_in[2];
  const float* Wq    = (const float*)d_in[3];
  const float* bq    = (const float*)d_in[4];
  const float* Wk    = (const float*)d_in[5];
  const float* bk    = (const float*)d_in[6];
  const float* Wv    = (const float*)d_in[7];
  const float* bv    = (const float*)d_in[8];
  const float* Wo    = (const float*)d_in[9];
  const float* bo    = (const float*)d_in[10];
  float* out = (float*)d_out;

  const int B = 2, S = 2048, H = 2048, Hq = 16, Hkv = 4, D = 128;
  const int M = B * S;      // 4096
  const int Nkv = Hkv * D;  // 512

  _Float16* q_h = (_Float16*)d_ws;        // [M, H]   f16
  _Float16* k_h = q_h + (size_t)M * H;    // [M, Nkv] f16
  _Float16* v_h = k_h + (size_t)M * Nkv;  // [M, Nkv] f16
  _Float16* c_h = v_h + (size_t)M * Nkv;  // [M, H]   f16  (~42MB total)

  dim3 blk(256);
  hipLaunchKernelGGL((gemm_bias_kernel<false, true>), dim3(H / 128, M / 128), blk, 0,
                     stream, (const void*)query, Wq, bq, (void*)q_h, M, H, H);
  hipLaunchKernelGGL((gemm_bias_kernel<false, true>), dim3(Nkv / 128, M / 128), blk, 0,
                     stream, (const void*)key_, Wk, bk, (void*)k_h, M, Nkv, H);
  hipLaunchKernelGGL((gemm_bias_kernel<false, true>), dim3(Nkv / 128, M / 128), blk, 0,
                     stream, (const void*)value, Wv, bv, (void*)v_h, M, Nkv, H);
  hipLaunchKernelGGL(gqa_attn_kernel, dim3(S / 128, B * Hq), blk, 0, stream,
                     q_h, k_h, v_h, c_h, S);
  hipLaunchKernelGGL((gemm_bias_kernel<true, false>), dim3(H / 128, M / 128), blk, 0,
                     stream, (const void*)c_h, Wo, bo, (void*)out, M, H, H);
}
